// TFLiteSTFT_15865609191757
// MI455X (gfx1250) — matrix operand order repaired and checked by reading, not compiled, not run on hardware
//
#include <hip/hip_runtime.h>
#include <hip/hip_bf16.h>

// ---------------------------------------------------------------------------
// Batched STFT as split-f16 WMMA GEMM for gfx1250 (MI455X, wave32).
//
//   C[b] (F x NJ) = A[b] (F x L) * B^T,  B is (NJ x L) row-major:
//     A[b][f][l]   = signals[b][f*STEP + l]            (window folded into B)
//     B[2k+0][l]   = window[l] * cos(-2*pi*k*l/L)
//     B[2k+1][l]   = window[l] * sin(-2*pi*k*l/L)
//   => C[b][f][:] is the interleaved (re,im) complex64 row of the output.
//
// Precision: x = hi(f16) + lo(f16);  D = Ah*Bh  (chain P)
//                                      + Ah*Bl + Al*Bh  (chain Q), fp32 acc.
// Two independent loop-carried WMMA chains per tile -> no intra-iteration
// C->D serialization, no accumulator copies, no hazard NOPs in the loop.
// ---------------------------------------------------------------------------

typedef __attribute__((ext_vector_type(16))) _Float16 v16h;
typedef __attribute__((ext_vector_type(8)))  _Float16 v8h;   // 16B = b128 load
typedef __attribute__((ext_vector_type(4)))  _Float16 v4h;
typedef __attribute__((ext_vector_type(8)))  float    v8f;

#define STFT_B      16
#define STFT_T      480000
#define STFT_L      800        // frame_length == fft_length
#define STFT_STEP   200
#define STFT_F      2397       // 1 + (T - L)/STEP
#define STFT_NJ     802        // 2 * (L/2 + 1) interleaved re/im rows
#define STFT_NJ_PAD 832        // 52 tiles of 16 -> 13 groups of 4 tiles
#define F_TILES     150        // ceil(2397/16)
#define FT_PER_BLK  8          // 8 waves per block, one f-tile each
#define N_GROUPS    13         // 832 / 64

// --- Kernel 1: windowed DFT basis, split into f16 hi/lo planes -------------
__global__ void stft_basis_kernel(const float* __restrict__ window,
                                  _Float16* __restrict__ bhi,
                                  _Float16* __restrict__ blo) {
    int idx = blockIdx.x * blockDim.x + threadIdx.x;
    if (idx >= STFT_NJ_PAD * STFT_L) return;
    int j = idx / STFT_L;
    int l = idx - j * STFT_L;
    float val = 0.0f;
    if (j < STFT_NJ) {
        int k = j >> 1;
        int r = (k * l) % STFT_L;            // exact reduction: angle in [-2pi, 0]
        float ang = -6.2831853071795864769f * (float)r * (1.0f / (float)STFT_L);
        float c = (j & 1) ? sinf(ang) : cosf(ang);
        val = c * window[l];
    }
    _Float16 h = (_Float16)val;
    bhi[idx] = h;
    blo[idx] = (_Float16)(val - (float)h);
}

// --- Kernel 2: split signals f32 -> f16 hi/lo planes (x4 vectorized) -------
__global__ void stft_split_kernel(const float* __restrict__ sig,
                                  _Float16* __restrict__ shi,
                                  _Float16* __restrict__ slo) {
    int q = blockIdx.x * blockDim.x + threadIdx.x;
    int base = q * 4;
    if (base >= STFT_B * STFT_T) return;
    float4 s = *(const float4*)(sig + base);
    v4h hi, lo;
    hi.x = (_Float16)s.x; lo.x = (_Float16)(s.x - (float)hi.x);
    hi.y = (_Float16)s.y; lo.y = (_Float16)(s.y - (float)hi.y);
    hi.z = (_Float16)s.z; lo.z = (_Float16)(s.z - (float)hi.z);
    hi.w = (_Float16)s.w; lo.w = (_Float16)(s.w - (float)hi.w);
    *(v4h*)(shi + base) = hi;
    *(v4h*)(slo + base) = lo;
}

// --- Kernel 3: split-f16 WMMA GEMM -----------------------------------------
// grid = (N_GROUPS, ceil(F_TILES/8), B); block = 256 (8 waves).
// Each wave: one 16(F) x 64(N) output strip = 4 tiles x 2 accumulator chains.
__global__ void __launch_bounds__(256)
stft_gemm_kernel(const _Float16* __restrict__ shi,
                 const _Float16* __restrict__ slo,
                 const _Float16* __restrict__ bhi,
                 const _Float16* __restrict__ blo,
                 float* __restrict__ out) {
    const int lane   = threadIdx.x & 31;
    const int wave   = threadIdx.x >> 5;
    const int sub    = lane & 15;         // M row (A) / N column (B,C)
    const int hiHalf = lane >> 4;         // upper half-wave -> upper K chunks

    const int fTile = blockIdx.y * FT_PER_BLK + wave;
    if (fTile >= F_TILES) return;         // wave-uniform: EXEC stays all-ones
    const int b      = blockIdx.z;
    const int nTile0 = blockIdx.x * 4;

    // Operand k order chosen here, same for A and B (16-bit, wave32):
    //  A 16x32: lanes 0-15 hold K 0..15, lanes 16-31 hold K 16..31 (contiguous)
    //  B 32x16: lanes 0-15 hold K 0..15, lanes 16-31 hold K 16..31 (contiguous)
    const int aOff0 = 16 * hiHalf;
    const int aOff1 = 16 * hiHalf + 8;
    const int bOff  = 16 * hiHalf;

    // A fragment rows: overlapping signal windows, clamp padded rows in-bounds.
    int f = fTile * 16 + sub;
    int fc = (f < STFT_F) ? f : (STFT_F - 1);
    const _Float16* arow_h = shi + (size_t)b * STFT_T + (size_t)fc * STFT_STEP;
    const _Float16* arow_l = slo + (size_t)b * STFT_T + (size_t)fc * STFT_STEP;

    // B fragment rows: basis row j = column N of the output tile.
    const _Float16* brow_h[4];
    const _Float16* brow_l[4];
#pragma unroll
    for (int t = 0; t < 4; ++t) {
        int j = (nTile0 + t) * 16 + sub;  // < 832 always; rows >= 802 are zero
        brow_h[t] = bhi + (size_t)j * STFT_L;
        brow_l[t] = blo + (size_t)j * STFT_L;
    }

    v8f accP[4] = {};   // Ah*Bh chain
    v8f accQ[4] = {};   // Ah*Bl + Al*Bh correction chain

#pragma unroll 5
    for (int k0 = 0; k0 < STFT_L; k0 += 32) {
        union { v16h v; v8h h[2]; } ah, al;
        ah.h[0] = *(const v8h*)(arow_h + k0 + aOff0);   // 16B-aligned b128
        ah.h[1] = *(const v8h*)(arow_h + k0 + aOff1);
        al.h[0] = *(const v8h*)(arow_l + k0 + aOff0);
        al.h[1] = *(const v8h*)(arow_l + k0 + aOff1);
#pragma unroll
        for (int t = 0; t < 4; ++t) {
            union { v16h v; v8h h[2]; } bh, bl;
            bh.h[0] = *(const v8h*)(brow_h[t] + k0 + bOff);
            bh.h[1] = *(const v8h*)(brow_h[t] + k0 + bOff + 8);
            bl.h[0] = *(const v8h*)(brow_l[t] + k0 + bOff);
            bl.h[1] = *(const v8h*)(brow_l[t] + k0 + bOff + 8);
            accP[t] = __builtin_amdgcn_wmma_f32_16x16x32_f16(
                false, ah.v, false, bh.v, (short)0, accP[t], false, false);
            accQ[t] = __builtin_amdgcn_wmma_f32_16x16x32_f16(
                false, ah.v, false, bl.v, (short)0, accQ[t], false, false);
            accQ[t] = __builtin_amdgcn_wmma_f32_16x16x32_f16(
                false, al.v, false, bh.v, (short)0, accQ[t], false, false);
        }
    }

    // Epilogue: sum the two chains and store.
    // C/D layout: VGPR v -> M = v + 8*hiHalf, N = sub. Row = interleaved re/im.
#pragma unroll
    for (int t = 0; t < 4; ++t) {
        int j = (nTile0 + t) * 16 + sub;
        if (j >= STFT_NJ) continue;
#pragma unroll
        for (int v = 0; v < 8; ++v) {
            int row = fTile * 16 + v + 8 * hiHalf;
            if (row < STFT_F)
                out[((size_t)b * STFT_F + row) * STFT_NJ + j] = accP[t][v] + accQ[t][v];
        }
    }
}

extern "C" void kernel_launch(void* const* d_in, const int* in_sizes, int n_in,
                              void* d_out, int out_size, void* d_ws, size_t ws_size,
                              hipStream_t stream) {
    const float* signals = (const float*)d_in[0];
    const float* window  = (const float*)d_in[1];
    // d_in[2..4] = frame_length/frame_step/fft_length scalars (fixed by setup).
    float* out = (float*)d_out;

    // Workspace layout (all offsets 256B-aligned by construction): ~33.4 MB.
    _Float16* bhi = (_Float16*)d_ws;
    _Float16* blo = bhi + (size_t)STFT_NJ_PAD * STFT_L;             // +1.33 MB
    _Float16* shi = blo + (size_t)STFT_NJ_PAD * STFT_L;             // +1.33 MB
    _Float16* slo = shi + (size_t)STFT_B * STFT_T;                  // +15.36 MB

    {   // basis: 832*800 elements
        int total = STFT_NJ_PAD * STFT_L;
        stft_basis_kernel<<<(total + 255) / 256, 256, 0, stream>>>(window, bhi, blo);
    }
    {   // signal split: 7.68M elements, 4 per thread
        int quads = (STFT_B * STFT_T) / 4;
        stft_split_kernel<<<(quads + 255) / 256, 256, 0, stream>>>(signals, shi, slo);
    }
    {
        dim3 grid(N_GROUPS, (F_TILES + FT_PER_BLK - 1) / FT_PER_BLK, STFT_B);
        stft_gemm_kernel<<<grid, 256, 0, stream>>>(shi, slo, bhi, blo, out);
    }
}
